// MixtureOfExperts_55018531062013
// MI455X (gfx1250) — compile-verified, matrix operand order checked by reading, not run on hardware
//
#include <hip/hip_runtime.h>
#include <math.h>

#define NE 8
#define NK 2
#define ND 1024
#define NF 4096
#define NT 4096
#define LNEPS 1e-5f
#define TT 16

typedef __attribute__((ext_vector_type(16))) __bf16 v16bf;
typedef __attribute__((ext_vector_type(8)))  __bf16 bf16x8;
typedef __attribute__((ext_vector_type(4)))  __bf16 bf16x4;
typedef __attribute__((ext_vector_type(8)))  float  v8f;

// LDS strides (padded to rotate banks: +16B per row for bf16, +16B for f32)
#define HS (NF + 8)                 // h tile row stride, bf16 elements (8208 B)
#define XS (ND + 8)                 // x tile row stride, bf16 elements (2064 B)
#define YS (ND + 4)                 // y tile row stride, f32 elements (4112 B)
#define HBYTES (TT * HS * 2)        // 131328
#define REGION2 (TT * YS * 4)       // 65792 (>= x tile's 33024; overlaid)
#define SMEM_BYTES (HBYTES + REGION2)

// ---------------------------------------------------------------- router
__global__ void k_router(const float* __restrict__ x, const float* __restrict__ Wr,
                         const float* __restrict__ br, float* __restrict__ gate,
                         int* __restrict__ cnt, int* __restrict__ list) {
  const int wv = threadIdx.x >> 5, lane = threadIdx.x & 31;
  const int t = blockIdx.x * 8 + wv;
  if (t >= NT) return;
  float acc[NE];
#pragma unroll
  for (int e = 0; e < NE; ++e) acc[e] = 0.f;
  const float* xr = x + (size_t)t * ND;
  for (int d = lane; d < ND; d += 32) {
    const float xv = xr[d];
    const float4* wr = (const float4*)(Wr + (size_t)d * NE);
    const float4 w0 = wr[0], w1 = wr[1];
    acc[0] += xv * w0.x; acc[1] += xv * w0.y; acc[2] += xv * w0.z; acc[3] += xv * w0.w;
    acc[4] += xv * w1.x; acc[5] += xv * w1.y; acc[6] += xv * w1.z; acc[7] += xv * w1.w;
  }
#pragma unroll
  for (int e = 0; e < NE; ++e)
#pragma unroll
    for (int off = 16; off; off >>= 1) acc[e] += __shfl_xor(acc[e], off, 32);
  if (lane == 0) {
    float p[NE], mx = -1e30f, s = 0.f;
#pragma unroll
    for (int e = 0; e < NE; ++e) { acc[e] += br[e]; mx = fmaxf(mx, acc[e]); }
#pragma unroll
    for (int e = 0; e < NE; ++e) { p[e] = __expf(acc[e] - mx); s += p[e]; }
    const float inv = 1.f / s;
#pragma unroll
    for (int e = 0; e < NE; ++e) p[e] *= inv;
    int i1 = 0;
#pragma unroll
    for (int e = 1; e < NE; ++e) if (p[e] > p[i1]) i1 = e;
    int i2 = (i1 == 0) ? 1 : 0;
#pragma unroll
    for (int e = 0; e < NE; ++e) if (e != i1 && p[e] > p[i2]) i2 = e;
    const float den = 1.f / (p[i1] + p[i2]);
#pragma unroll
    for (int e = 0; e < NE; ++e)
      gate[(size_t)t * NE + e] = (e == i1) ? p[i1] * den : ((e == i2) ? p[i2] * den : 0.f);
    const int p1 = atomicAdd(&cnt[i1], 1); list[i1 * NT + p1] = t * NK + 0;
    const int p2 = atomicAdd(&cnt[i2], 1); list[i2 * NT + p2] = t * NK + 1;
  }
}

// ---------------------------------------------------------------- converts
__global__ void k_cvt_x(const float* __restrict__ x, __bf16* __restrict__ xb, long n4) {
  const long i = (long)blockIdx.x * blockDim.x + threadIdx.x;
  if (i >= n4) return;
  const float4 v = ((const float4*)x)[i];
  bf16x4 o;
  o[0] = (__bf16)v.x; o[1] = (__bf16)v.y; o[2] = (__bf16)v.z; o[3] = (__bf16)v.w;
  ((bf16x4*)xb)[i] = o;
}

// src: [z][R][C] f32  ->  dst: [z][C][R] bf16   (block = 32x8)
__global__ void k_transpose_cvt(const float* __restrict__ src, __bf16* __restrict__ dst,
                                int R, int C) {
  __shared__ float tile[32][33];
  const size_t base = (size_t)blockIdx.z * R * C;
  const int c0 = blockIdx.x * 32, r0 = blockIdx.y * 32;
#pragma unroll
  for (int i = 0; i < 4; ++i) {
    const int r = r0 + threadIdx.y + i * 8;
    tile[threadIdx.y + i * 8][threadIdx.x] = src[base + (size_t)r * C + c0 + threadIdx.x];
  }
  __syncthreads();
#pragma unroll
  for (int i = 0; i < 4; ++i) {
    const int c = c0 + threadIdx.y + i * 8;
    dst[base + (size_t)c * R + r0 + threadIdx.x] = (__bf16)tile[threadIdx.x][threadIdx.y + i * 8];
  }
}

// ---------------------------------------------------------------- fragment load
// 16-bit A/B layout: lane<16 -> K-chunks {0..7, 16..23}+kbase; lane>=16 -> {8..15, 24..31}
__device__ __forceinline__ v16bf load_frag16(const __bf16* rowbase, int k0) {
  const bf16x8 lo = *(const bf16x8*)(rowbase + k0);
  const bf16x8 hi = *(const bf16x8*)(rowbase + k0 + 16);
  return __builtin_shufflevector(lo, hi, 0, 1, 2, 3, 4, 5, 6, 7,
                                 8, 9, 10, 11, 12, 13, 14, 15);
}

// ---------------------------------------------------------------- fused expert kernel
__global__ void __launch_bounds__(256, 1)
k_moe(const __bf16* __restrict__ xb, const __bf16* __restrict__ W1T,
      const __bf16* __restrict__ W2T,
      const float* __restrict__ b1, const float* __restrict__ g1, const float* __restrict__ be1,
      const float* __restrict__ b2, const float* __restrict__ g2, const float* __restrict__ be2,
      const float* __restrict__ gate, const int* __restrict__ cnt, const int* __restrict__ list,
      float* __restrict__ yscr) {
  extern __shared__ unsigned char smem[];
  __bf16* hT = (__bf16*)smem;                  // [TT][HS] bf16
  __bf16* xT = (__bf16*)(smem + HBYTES);       // [TT][XS] bf16 (phase A)
  float*  yT = (float*)(smem + HBYTES);        // [TT][YS] f32  (phase B, overlays xT)
  __shared__ int slotS[TT];

  const int e = blockIdx.y;
  const int nTok = cnt[e];
  const int t0 = blockIdx.x * TT;
  if (t0 >= nTok) return;                      // static grid, dynamic work (top-k sparsity)

  const int tid = threadIdx.x, lane = tid & 31, wv = tid >> 5;

  if (tid < TT)
    slotS[tid] = (t0 + tid < nTok) ? list[e * NT + t0 + tid] : -1;
  __syncthreads();

  // ---- stage compacted x rows (bf16) into LDS, 16B chunks
  {
    const int cpr = ND / 8;  // 128 chunks per row
    for (int i = tid; i < TT * cpr; i += 256) {
      const int row = i / cpr, c = i % cpr;
      const int s = slotS[row];
      const int tok = (s >= 0) ? (s / NK) : (slotS[0] / NK);  // clamp pad rows (row0 valid)
      *(bf16x8*)(xT + row * XS + c * 8) = *(const bf16x8*)(xb + (size_t)tok * ND + c * 8);
    }
  }
  __syncthreads();

  // ---- Phase A: h = relu(x @ W1 + b1) -> bf16 LDS
  {
    const int r = lane & 15, hb = (lane >> 4) * 8;
    const __bf16* aRow = xT + r * XS;
    for (int nt = wv; nt < NF / 16; nt += 8) {
      const int f0 = nt * 16;
      const __bf16* bRow = W1T + ((size_t)e * NF + f0 + r) * ND;
      v8f acc = {};
#pragma unroll 4
      for (int ko = 0; ko < ND / 32; ++ko) {
        const int k = ko * 32 + hb;
        if ((ko & 7) == 0) __builtin_prefetch(bRow + ko * 32 + 256, 0, 0);
        const v16bf a = load_frag16(aRow, k);
        const v16bf b = load_frag16(bRow, k);
        acc = __builtin_amdgcn_wmma_f32_16x16x32_bf16(false, a, false, b,
                                                      (short)0, acc, false, false);
      }
      const int n = lane & 15, mb = (lane >> 4) * 8;
      const float bias = b1[(size_t)e * NF + f0 + n];
#pragma unroll
      for (int i = 0; i < 8; ++i) {
        float v = acc[i] + bias;
        hT[(mb + i) * HS + f0 + n] = (__bf16)(v > 0.f ? v : 0.f);
      }
    }
  }
  __syncthreads();

  // ---- Phase A2: LayerNorm over F per token, in place (*g1 + be1)
  for (int t = wv; t < TT; t += 8) {
    __bf16* hr = hT + t * HS;
    float s = 0.f, s2 = 0.f;
    for (int j = lane; j < NF; j += 32) { const float v = (float)hr[j]; s += v; s2 += v * v; }
#pragma unroll
    for (int off = 16; off; off >>= 1) { s += __shfl_xor(s, off, 32); s2 += __shfl_xor(s2, off, 32); }
    const float mu = s / NF;
    const float rs = rsqrtf(fmaxf(s2 / NF - mu * mu, 0.f) + LNEPS);
    for (int j = lane; j < NF; j += 32) {
      const float v = ((float)hr[j] - mu) * rs * g1[(size_t)e * NF + j] + be1[(size_t)e * NF + j];
      hr[j] = (__bf16)v;
    }
  }
  __syncthreads();

  // ---- Phase B: y = relu(hln @ W2 + b2) -> f32 LDS (8 output tiles per wave)
  {
    const int r = lane & 15, hb = (lane >> 4) * 8;
    const __bf16* aRow = hT + r * HS;
    v8f acc[8] = {};
    const __bf16* bRow[8];
#pragma unroll
    for (int it = 0; it < 8; ++it)
      bRow[it] = W2T + ((size_t)e * ND + (wv * 8 + it) * 16 + r) * NF;
    for (int ko = 0; ko < NF / 32; ++ko) {
      const int k = ko * 32 + hb;
      const v16bf a = load_frag16(aRow, k);
#pragma unroll
      for (int it = 0; it < 8; ++it) {
        const v16bf b = load_frag16(bRow[it], k);
        acc[it] = __builtin_amdgcn_wmma_f32_16x16x32_bf16(false, a, false, b,
                                                          (short)0, acc[it], false, false);
      }
    }
    const int n = lane & 15, mb = (lane >> 4) * 8;
#pragma unroll
    for (int it = 0; it < 8; ++it) {
      const int d0 = (wv * 8 + it) * 16;
      const float bias = b2[(size_t)e * ND + d0 + n];
#pragma unroll
      for (int i = 0; i < 8; ++i) {
        const float v = acc[it][i] + bias;
        yT[(mb + i) * YS + d0 + n] = v > 0.f ? v : 0.f;
      }
    }
  }
  __syncthreads();

  // ---- Phase B2: LayerNorm over D, gate-scale, write unique slot row (no atomics)
  for (int t = wv; t < TT; t += 8) {
    const int slot = slotS[t];
    if (slot < 0) continue;
    const int tok = slot / NK;
    const float gw = gate[(size_t)tok * NE + e];
    const float* yr = yT + t * YS;
    float s = 0.f, s2 = 0.f;
    for (int j = lane; j < ND; j += 32) { const float v = yr[j]; s += v; s2 += v * v; }
#pragma unroll
    for (int off = 16; off; off >>= 1) { s += __shfl_xor(s, off, 32); s2 += __shfl_xor(s2, off, 32); }
    const float mu = s / ND;
    const float rs = rsqrtf(fmaxf(s2 / ND - mu * mu, 0.f) + LNEPS);
    float* orow = yscr + (size_t)slot * ND;
    for (int j = lane; j < ND; j += 32) {
      const float v = (yr[j] - mu) * rs * g2[(size_t)e * ND + j] + be2[(size_t)e * ND + j];
      orow[j] = gw * v;
    }
  }
}

// ---------------------------------------------------------------- combine
__global__ void k_combine(const float* __restrict__ yscr, float* __restrict__ out, long n) {
  const long i = (long)blockIdx.x * blockDim.x + threadIdx.x;
  if (i >= n) return;
  const long t = i / ND, j = i % ND;
  out[i] = yscr[(size_t)(2 * t) * ND + j] + yscr[(size_t)(2 * t + 1) * ND + j];
}

// ---------------------------------------------------------------- host
extern "C" void kernel_launch(void* const* d_in, const int* in_sizes, int n_in,
                              void* d_out, int out_size, void* d_ws, size_t ws_size,
                              hipStream_t stream) {
  (void)in_sizes; (void)n_in; (void)out_size; (void)ws_size;
  const float* x  = (const float*)d_in[0];
  const float* Wr = (const float*)d_in[1];
  const float* br = (const float*)d_in[2];
  const float* W1 = (const float*)d_in[3];
  const float* b1 = (const float*)d_in[4];
  const float* g1 = (const float*)d_in[5];
  const float* be1= (const float*)d_in[6];
  const float* W2 = (const float*)d_in[7];
  const float* b2 = (const float*)d_in[8];
  const float* g2 = (const float*)d_in[9];
  const float* be2= (const float*)d_in[10];

  unsigned char* ws = (unsigned char*)d_ws;
  size_t off = 0;
  auto carve = [&](size_t bytes) { size_t o = off; off = (off + bytes + 255) & ~(size_t)255; return o; };
  float*  gate = (float*)(ws + carve((size_t)NT * NE * 4));
  int*    cnt  = (int*)  (ws + carve((size_t)NE * 4));
  int*    list = (int*)  (ws + carve((size_t)NE * NT * 4));
  __bf16* xb   = (__bf16*)(ws + carve((size_t)NT * ND * 2));
  __bf16* W1T  = (__bf16*)(ws + carve((size_t)NE * NF * ND * 2));
  __bf16* W2T  = (__bf16*)(ws + carve((size_t)NE * ND * NF * 2));
  float*  yscr = (float*)(ws + carve((size_t)NT * NK * ND * 4));

  hipMemsetAsync(cnt, 0, NE * sizeof(int), stream);
  k_router<<<NT / 8, 256, 0, stream>>>(x, Wr, br, gate, cnt, list);
  k_cvt_x<<<(NT * ND / 4 + 255) / 256, 256, 0, stream>>>(x, xb, (long)NT * ND / 4);
  dim3 tb(32, 8);
  k_transpose_cvt<<<dim3(NF / 32, ND / 32, NE), tb, 0, stream>>>(W1, W1T, ND, NF);
  k_transpose_cvt<<<dim3(ND / 32, NF / 32, NE), tb, 0, stream>>>(W2, W2T, NF, ND);

  hipFuncSetAttribute(reinterpret_cast<const void*>(k_moe),
                      hipFuncAttributeMaxDynamicSharedMemorySize, SMEM_BYTES);
  k_moe<<<dim3(NT / TT, NE), 256, SMEM_BYTES, stream>>>(
      xb, W1T, W2T, b1, g1, be1, b2, g2, be2, gate, cnt, list, yscr);

  k_combine<<<((long)NT * ND + 255) / 256, 256, 0, stream>>>(yscr, (float*)d_out, (long)NT * ND);
}
